// HyperGraphTransformer_30468497998155
// MI455X (gfx1250) — matrix operand order repaired and checked by reading, not compiled, not run on hardware
//
#include <hip/hip_runtime.h>
#include <hip/hip_bf16.h>

// ---------------------------------------------------------------------------
// CDNA5 (gfx1250) HGT forward: typed linears via v_wmma_f32_16x16x32_f16,
// edge attention via gather/atomic passes, LayerNorm via wave32 shuffles.
// ---------------------------------------------------------------------------

typedef __attribute__((ext_vector_type(16))) _Float16 v16h;
typedef __attribute__((ext_vector_type(8)))  _Float16 v8h;
typedef __attribute__((ext_vector_type(8)))  float    v8f;

#define LDS_ST 136   // 128 + 8 halves pad: row stride 272B (16B aligned, bank-rotating)

// ---------------- weight prep kernels --------------------------------------

// Wq2[t,d,o] = Wq[t,d,o] * rpri[o/16] / 4 ; bias likewise (row r==128 handles bias)
__global__ void fuse_scale_q(const float* __restrict__ W, const float* __restrict__ bsrc,
                             const float* __restrict__ rpri,
                             float* __restrict__ W2, float* __restrict__ b2, int T)
{
    long idx = (long)blockIdx.x * 256 + threadIdx.x;
    long tot = (long)T * 129 * 128;
    if (idx >= tot) return;
    int o = (int)(idx % 128);
    long rr = idx / 128;
    int r = (int)(rr % 129);
    int t = (int)(rr / 129);
    float sc = rpri[o >> 4] * 0.25f;
    if (r < 128)
        W2[((long)t * 128 + r) * 128 + o] = W[((long)t * 128 + r) * 128 + o] * sc;
    else
        b2[(long)t * 128 + o] = bsrc[(long)t * 128 + o] * sc;
}

// Wk2[t,d,h*16+j] = sum_i Wk[t,d,h*16+i] * R[h,i,j] ; bias likewise (r==128)
__global__ void fuse_headmat(const float* __restrict__ W, const float* __restrict__ bsrc,
                             const float* __restrict__ R,
                             float* __restrict__ W2, float* __restrict__ b2, int T)
{
    long idx = (long)blockIdx.x * 256 + threadIdx.x;
    long tot = (long)T * 129 * 128;
    if (idx >= tot) return;
    int o = (int)(idx % 128);
    long rr = idx / 128;
    int r = (int)(rr % 129);
    int t = (int)(rr / 129);
    int h = o >> 4, j = o & 15;
    const float* Rh = R + (long)h * 256 + j;
    float acc = 0.f;
    if (r < 128) {
        const float* wrow = W + ((long)t * 128 + r) * 128 + h * 16;
        #pragma unroll
        for (int i = 0; i < 16; ++i) acc += wrow[i] * Rh[i * 16];
        W2[((long)t * 128 + r) * 128 + o] = acc;
    } else {
        const float* brow = bsrc + (long)t * 128 + h * 16;
        #pragma unroll
        for (int i = 0; i < 16; ++i) acc += brow[i] * Rh[i * 16];
        b2[(long)t * 128 + o] = acc;
    }
}

// Pack f32 [T][128 d][128 o] row-major into WMMA-B fragment order (f16):
// out[(((t*8+col)*4+ks)*32+lane)*16 + j] ; lane holds column o=col*16+(lane&15),
// K = ks*32 + 16*(lane>=16) + j  (contiguous K, same order as the A fragment: lanes 0-15 K=j, lanes 16-31 K=16+j)
__global__ void pack_half(const float* __restrict__ W, _Float16* __restrict__ out, int T)
{
    long idx = (long)blockIdx.x * 256 + threadIdx.x;
    long tot = (long)T * 16384;
    if (idx >= tot) return;
    int j    = (int)(idx & 15);
    long r   = idx >> 4;
    int lane = (int)(r & 31);  r >>= 5;
    int ks   = (int)(r & 3);   r >>= 2;
    int col  = (int)(r & 7);
    int t    = (int)(r >> 3);
    int k = ks * 32 + ((lane >> 4) & 1) * 16 + j;
    int o = col * 16 + (lane & 15);
    out[idx] = (_Float16)W[(long)t * 16384 + (long)k * 128 + o];
}

// ---------------- WMMA typed GEMM ------------------------------------------
// out[n,:] = act(x[n,:]) @ W[type[n]] + bias[type[n]]   (D=128, 16-node tiles)
// act: 0 none, 1 gelu(tanh), 2 relu   (applied to the INPUT x)
__global__ __launch_bounds__(256) void typed_gemm16(
    const float* __restrict__ x, const int* __restrict__ types,
    const _Float16* __restrict__ Wpk, const float* __restrict__ bias,
    float* __restrict__ out, int N, int T, int act)
{
    __shared__ _Float16 sx[16 * LDS_ST];
    __shared__ int sty[16];
    __shared__ unsigned smask;

    const int tile = blockIdx.x;
    const int tid  = threadIdx.x;
    if (tid == 0) smask = 0u;
    __syncthreads();

    // stage 16x128 input tile into LDS as f16 (activation applied)
    #pragma unroll
    for (int i = 0; i < 8; ++i) {
        int elem = tid + i * 256;
        int r = elem >> 7, c = elem & 127;
        int g = tile * 16 + r;
        float v = 0.f;
        if (g < N) v = x[(long)g * 128 + c];
        if (act == 1) {
            float v3 = v * v * v;
            v = 0.5f * v * (1.f + tanhf(0.7978845608f * (v + 0.044715f * v3)));
        } else if (act == 2) {
            v = fmaxf(v, 0.f);
        }
        sx[r * LDS_ST + c] = (_Float16)v;
    }
    if (tid < 16) {
        int g = tile * 16 + tid;
        int ty = (g < N) ? (types ? types[g] : 0) : -1;
        sty[tid] = ty;
        if (ty >= 0) atomicOr(&smask, 1u << ty);
    }
    __syncthreads();

    const int wave = tid >> 5;          // column tile 0..7
    const int lane = tid & 31;
    const int hsel = (lane >> 4) & 1;
    const int row  = lane & 15;
    const int colg = wave * 16 + row;   // output column for this lane
    const unsigned mask = smask;

    // Hoist this lane's 8 writeback row types + bounds out of the type loop:
    // lane writes rows r = c + 8*hsel, column colg.
    int  myty[8];
    bool gok[8];
    #pragma unroll
    for (int c = 0; c < 8; ++c) {
        int r = c + hsel * 8;
        myty[c] = sty[r];
        gok[c]  = (tile * 16 + r) < N;
    }
    float* orow0 = out + ((long)(tile * 16) + hsel * 8) * 128 + colg;

    for (int tt = 0; tt < T; ++tt) {
        if (T > 1 && !((mask >> tt) & 1u)) continue;   // block-uniform branch
        float bcol = bias[(long)tt * 128 + colg];
        v8f acc;
        #pragma unroll
        for (int c = 0; c < 8; ++c) acc[c] = bcol;

        const _Float16* wp = Wpk + ((long)tt * 8 + wave) * 2048;
        #pragma unroll
        for (int ks = 0; ks < 4; ++ks) {
            // A fragment: row=lane&15, K = ks*32 + 16*hsel + {0..15}
            int kb = ks * 32 + hsel * 16;
            v8h lo = *reinterpret_cast<const v8h*>(&sx[row * LDS_ST + kb]);
            v8h hi = *reinterpret_cast<const v8h*>(&sx[row * LDS_ST + kb + 8]);
            v16h a;
            #pragma unroll
            for (int j = 0; j < 8; ++j) { a[j] = lo[j]; a[j + 8] = hi[j]; }
            // B fragment: pre-packed, contiguous 32B per lane
            v16h b = *reinterpret_cast<const v16h*>(wp + ((long)ks * 32 + lane) * 16);
            acc = __builtin_amdgcn_wmma_f32_16x16x32_f16(
                false, a, false, b, (short)0, acc, false, false);
        }
        // D layout: acc[c] = C[c + 8*hsel][colg]; write only rows of type tt
        #pragma unroll
        for (int c = 0; c < 8; ++c) {
            if (gok[c] && (T == 1 || myty[c] == tt))
                orow0[(long)c * 128] = acc[c];
        }
    }
}

// ---------------- edge attention kernels -----------------------------------

__device__ __forceinline__ unsigned pun_f32(float f) {
    unsigned u = __float_as_uint(f);
    return (u & 0x80000000u) ? ~u : (u | 0x80000000u);
}
__device__ __forceinline__ float unpun_f32(unsigned u) {
    unsigned b = (u & 0x80000000u) ? (u & 0x7FFFFFFFu) : ~u;
    return __uint_as_float(b);
}

__global__ void fill_u32(unsigned* __restrict__ p, unsigned v, long n)
{
    long i = (long)blockIdx.x * 256 + threadIdx.x;
    if (i < n) p[i] = v;
}

// pass 1: score[e,h] = q2[tgt]·k2[src] (rpri/sqrt(DK) pre-folded); segment max
__global__ void edge_score(const float* __restrict__ q2, const float* __restrict__ k2,
                           const int* __restrict__ src, const int* __restrict__ tgt,
                           float* __restrict__ score, unsigned* __restrict__ mbuf, long EH)
{
    long idx = (long)blockIdx.x * 256 + threadIdx.x;
    if (idx >= EH) return;
    long e = idx >> 3; int h = (int)(idx & 7);
    int sn = src[e], tn = tgt[e];
    const float4* qp = (const float4*)(q2 + (long)tn * 128 + h * 16);
    const float4* kp = (const float4*)(k2 + (long)sn * 128 + h * 16);
    float acc = 0.f;
    #pragma unroll
    for (int i = 0; i < 4; ++i) {
        float4 a = qp[i], b = kp[i];
        acc += a.x * b.x + a.y * b.y + a.z * b.z + a.w * b.w;
    }
    score[idx] = acc;
    atomicMax(mbuf + (long)tn * 8 + h, pun_f32(acc));
}

// pass 2: ex = exp(score - m[tgt]); segment sum
__global__ void edge_exp(float* __restrict__ score, const unsigned* __restrict__ mbuf,
                         float* __restrict__ sbuf, const int* __restrict__ tgt, long EH)
{
    long idx = (long)blockIdx.x * 256 + threadIdx.x;
    if (idx >= EH) return;
    long e = idx >> 3; int h = (int)(idx & 7);
    int tn = tgt[e];
    float m = unpun_f32(mbuf[(long)tn * 8 + h]);
    float ex = __expf(score[idx] - m);
    score[idx] = ex;
    atomicAdd(sbuf + (long)tn * 8 + h, ex);
}

// pass 3: agg[tgt,d] += ex/(s+eps) * v2[src,d]   (one thread per (e,d))
__global__ void edge_scatter(const float* __restrict__ score, const float* __restrict__ sbuf,
                             const float* __restrict__ v2,
                             const int* __restrict__ src, const int* __restrict__ tgt,
                             float* __restrict__ agg, long ED)
{
    long idx = (long)blockIdx.x * 256 + threadIdx.x;
    if (idx >= ED) return;
    long e = idx >> 7; int d = (int)(idx & 127); int h = d >> 4;
    int tn = tgt[e], sn = src[e];
    float w = score[e * 8 + h] / (sbuf[(long)tn * 8 + h] + 1e-9f);
    atomicAdd(agg + (long)tn * 128 + d, w * v2[(long)sn * 128 + d]);
}

// ---------------- gated skip + LayerNorm (one wave32 per node) --------------
__global__ __launch_bounds__(256) void node_finalize(
    const float* __restrict__ lin, const float* __restrict__ qf,
    const int* __restrict__ types, const float* __restrict__ skip,
    const float* __restrict__ lng, const float* __restrict__ lnb,
    float* __restrict__ out, int N)
{
    int node = blockIdx.x * 8 + (threadIdx.x >> 5);
    int lane = threadIdx.x & 31;
    if (node >= N) return;
    int t = types[node];
    float alpha = 1.f / (1.f + __expf(-skip[t]));
    long base = (long)node * 128 + lane * 4;
    float y[4];
    #pragma unroll
    for (int i = 0; i < 4; ++i)
        y[i] = lin[base + i] * alpha + qf[base + i] * (1.f - alpha);
    float s = y[0] + y[1] + y[2] + y[3];
    #pragma unroll
    for (int off = 16; off; off >>= 1) s += __shfl_xor(s, off, 32);
    float mu = s * (1.f / 128.f);
    float vs = 0.f;
    #pragma unroll
    for (int i = 0; i < 4; ++i) { float d = y[i] - mu; vs += d * d; }
    #pragma unroll
    for (int off = 16; off; off >>= 1) vs += __shfl_xor(vs, off, 32);
    float inv = rsqrtf(vs * (1.f / 128.f) + 1e-5f);
    #pragma unroll
    for (int i = 0; i < 4; ++i) {
        int c = lane * 4 + i;
        out[base + i] = (y[i] - mu) * inv * lng[c] + lnb[c];
    }
}

// ---------------------------------------------------------------------------

static inline long cdivl(long a, long b) { return (a + b - 1) / b; }

struct LayerParams {
    const _Float16 *pkq, *pkk, *pkv, *pka;
    const float *bq2, *bk2, *bv2, *ba;
    const float *skip, *lng, *lnb;
};

static void run_hgt_layer(const float* qfeat, const float* kvfeat,
                          const int* node_type, const LayerParams& P,
                          const int* esrc, const int* etgt, int N, int E, int T,
                          float* q2, float* k2, float* v2, float* agg,
                          float* linbuf, float* scoreBuf, unsigned* mbuf, float* sbuf,
                          float* outp, hipStream_t stream)
{
    long tiles = cdivl(N, 16);
    typed_gemm16<<<(unsigned)tiles, 256, 0, stream>>>(qfeat, node_type, P.pkq, P.bq2, q2, N, T, 0);
    typed_gemm16<<<(unsigned)tiles, 256, 0, stream>>>(kvfeat, node_type, P.pkk, P.bk2, k2, N, T, 0);
    typed_gemm16<<<(unsigned)tiles, 256, 0, stream>>>(kvfeat, node_type, P.pkv, P.bv2, v2, N, T, 0);

    long NH = (long)N * 8, EH = (long)E * 8, ED = (long)E * 128, ND = (long)N * 128;
    fill_u32<<<(unsigned)cdivl(NH, 256), 256, 0, stream>>>(mbuf, 0x007FFFFFu, NH);  // pun(-inf)
    fill_u32<<<(unsigned)cdivl(NH, 256), 256, 0, stream>>>((unsigned*)sbuf, 0u, NH);
    fill_u32<<<(unsigned)cdivl(ND, 256), 256, 0, stream>>>((unsigned*)agg, 0u, ND);

    edge_score<<<(unsigned)cdivl(EH, 256), 256, 0, stream>>>(q2, k2, esrc, etgt, scoreBuf, mbuf, EH);
    edge_exp<<<(unsigned)cdivl(EH, 256), 256, 0, stream>>>(scoreBuf, mbuf, sbuf, etgt, EH);
    edge_scatter<<<(unsigned)cdivl(ED, 256), 256, 0, stream>>>(scoreBuf, sbuf, v2, esrc, etgt, agg, ED);

    typed_gemm16<<<(unsigned)tiles, 256, 0, stream>>>(agg, node_type, P.pka, P.ba, linbuf, N, T, 1);
    node_finalize<<<(unsigned)cdivl(N, 8), 256, 0, stream>>>(linbuf, qfeat, node_type,
                                                             P.skip, P.lng, P.lnb, outp, N);
}

extern "C" void kernel_launch(void* const* d_in, const int* in_sizes, int n_in,
                              void* d_out, int out_size, void* d_ws, size_t ws_size,
                              hipStream_t stream)
{
    const float* node_feat = (const float*)d_in[0];
    const int*   node_type = (const int*)d_in[1];
    const int*   edge_pair = (const int*)d_in[2];
    const int N = in_sizes[1];
    const int E = in_sizes[3];
    const int T = in_sizes[15];          // a_skip has length T
    const int* src = edge_pair;
    const int* tgt = edge_pair + E;

    // a_: 4..17   b_: 18..31   mlp: 32..35
    const float* a_Wq  = (const float*)d_in[4];  const float* a_bq  = (const float*)d_in[5];
    const float* a_Wk  = (const float*)d_in[6];  const float* a_bk  = (const float*)d_in[7];
    const float* a_Wv  = (const float*)d_in[8];  const float* a_bv  = (const float*)d_in[9];
    const float* a_Wa  = (const float*)d_in[10]; const float* a_ba  = (const float*)d_in[11];
    const float* a_ratt= (const float*)d_in[12]; const float* a_rmsg= (const float*)d_in[13];
    const float* a_rpri= (const float*)d_in[14]; const float* a_skip= (const float*)d_in[15];
    const float* a_lng = (const float*)d_in[16]; const float* a_lnb = (const float*)d_in[17];
    const float* b_Wq  = (const float*)d_in[18]; const float* b_bq  = (const float*)d_in[19];
    const float* b_Wk  = (const float*)d_in[20]; const float* b_bk  = (const float*)d_in[21];
    const float* b_Wv  = (const float*)d_in[22]; const float* b_bv  = (const float*)d_in[23];
    const float* b_Wa  = (const float*)d_in[24]; const float* b_ba  = (const float*)d_in[25];
    const float* b_ratt= (const float*)d_in[26]; const float* b_rmsg= (const float*)d_in[27];
    const float* b_rpri= (const float*)d_in[28]; const float* b_skip= (const float*)d_in[29];
    const float* b_lng = (const float*)d_in[30]; const float* b_lnb = (const float*)d_in[31];
    const float* m_W1  = (const float*)d_in[32]; const float* m_b1  = (const float*)d_in[33];
    const float* m_W2  = (const float*)d_in[34]; const float* m_b2  = (const float*)d_in[35];

    // ---- carve workspace ----
    size_t off = 0;
    auto carve = [&](size_t bytes) -> void* {
        void* p = (char*)d_ws + off;
        off += (bytes + 255) & ~(size_t)255;
        return p;
    };
    size_t NB = (size_t)N * 128 * sizeof(float);
    float* B0 = (float*)carve(NB);
    float* B1 = (float*)carve(NB);
    float* B2 = (float*)carve(NB);
    float* B3 = (float*)carve(NB);
    float* B4 = (float*)carve(NB);
    float* B5 = (float*)carve(NB);
    float*    scoreBuf = (float*)carve((size_t)E * 8 * sizeof(float));
    unsigned* mbuf     = (unsigned*)carve((size_t)N * 8 * sizeof(unsigned));
    float*    sbuf     = (float*)carve((size_t)N * 8 * sizeof(float));
    float*    wf       = (float*)carve((size_t)T * 128 * 128 * sizeof(float));
    float* bq2a = (float*)carve((size_t)T * 128 * 4);
    float* bk2a = (float*)carve((size_t)T * 128 * 4);
    float* bv2a = (float*)carve((size_t)T * 128 * 4);
    float* bq2b = (float*)carve((size_t)T * 128 * 4);
    float* bk2b = (float*)carve((size_t)T * 128 * 4);
    float* bv2b = (float*)carve((size_t)T * 128 * 4);
    size_t PKB = (size_t)T * 16384 * sizeof(_Float16);
    _Float16* pk_qa = (_Float16*)carve(PKB);
    _Float16* pk_ka = (_Float16*)carve(PKB);
    _Float16* pk_va = (_Float16*)carve(PKB);
    _Float16* pk_aa = (_Float16*)carve(PKB);
    _Float16* pk_qb = (_Float16*)carve(PKB);
    _Float16* pk_kb = (_Float16*)carve(PKB);
    _Float16* pk_vb = (_Float16*)carve(PKB);
    _Float16* pk_ab = (_Float16*)carve(PKB);
    _Float16* pk_m1 = (_Float16*)carve((size_t)16384 * sizeof(_Float16));
    _Float16* pk_m2 = (_Float16*)carve((size_t)16384 * sizeof(_Float16));
    (void)ws_size; (void)n_in; (void)out_size;

    // ---- weight prep: fold rpri into Wq, ratt into Wk, rmsg into Wv; pack f16 ----
    long TW  = (long)T * 129 * 128;      // includes bias row
    long TPK = (long)T * 16384;
    unsigned gF = (unsigned)cdivl(TW, 256), gP = (unsigned)cdivl(TPK, 256);
    unsigned gP1 = (unsigned)cdivl(16384, 256);

    fuse_scale_q<<<gF, 256, 0, stream>>>(a_Wq, a_bq, a_rpri, wf, bq2a, T);
    pack_half   <<<gP, 256, 0, stream>>>(wf, pk_qa, T);
    fuse_headmat<<<gF, 256, 0, stream>>>(a_Wk, a_bk, a_ratt, wf, bk2a, T);
    pack_half   <<<gP, 256, 0, stream>>>(wf, pk_ka, T);
    fuse_headmat<<<gF, 256, 0, stream>>>(a_Wv, a_bv, a_rmsg, wf, bv2a, T);
    pack_half   <<<gP, 256, 0, stream>>>(wf, pk_va, T);
    pack_half   <<<gP, 256, 0, stream>>>(a_Wa, pk_aa, T);

    fuse_scale_q<<<gF, 256, 0, stream>>>(b_Wq, b_bq, b_rpri, wf, bq2b, T);
    pack_half   <<<gP, 256, 0, stream>>>(wf, pk_qb, T);
    fuse_headmat<<<gF, 256, 0, stream>>>(b_Wk, b_bk, b_ratt, wf, bk2b, T);
    pack_half   <<<gP, 256, 0, stream>>>(wf, pk_kb, T);
    fuse_headmat<<<gF, 256, 0, stream>>>(b_Wv, b_bv, b_rmsg, wf, bv2b, T);
    pack_half   <<<gP, 256, 0, stream>>>(wf, pk_vb, T);
    pack_half   <<<gP, 256, 0, stream>>>(b_Wa, pk_ab, T);

    pack_half<<<gP1, 256, 0, stream>>>(m_W1, pk_m1, 1);
    pack_half<<<gP1, 256, 0, stream>>>(m_W2, pk_m2, 1);

    // ---- layer a: self attention over edges (src -> tgt) ----
    LayerParams Pa{pk_qa, pk_ka, pk_va, pk_aa, bq2a, bk2a, bv2a, a_ba, a_skip, a_lng, a_lnb};
    float* nw = B5;
    run_hgt_layer(node_feat, node_feat, node_type, Pa, src, tgt, N, E, T,
                  B0, B1, B2, B3, B4, scoreBuf, mbuf, sbuf, nw, stream);

    // ---- MLP: he = relu(nw @ W1 + b1) @ W2 + b2 ----
    long tiles = cdivl(N, 16);
    float* h1 = B3;   // agg of layer a, free now
    float* he = B0;   // q2 of layer a, free now
    typed_gemm16<<<(unsigned)tiles, 256, 0, stream>>>(nw, nullptr, pk_m1, m_b1, h1, N, 1, 0);
    typed_gemm16<<<(unsigned)tiles, 256, 0, stream>>>(h1, nullptr, pk_m2, m_b2, he, N, 1, 2);

    // ---- layer b: edges reversed (tgt -> src), kv from he, q/skip from node_feat ----
    LayerParams Pb{pk_qb, pk_kb, pk_vb, pk_ab, bq2b, bk2b, bv2b, b_ba, b_skip, b_lng, b_lnb};
    run_hgt_layer(node_feat, he, node_type, Pb, tgt, src, N, E, T,
                  B1, B2, B3, B4, B5, scoreBuf, mbuf, sbuf, (float*)d_out, stream);
}
